// FeatureFBPConnection_32169305047400
// MI455X (gfx1250) — matrix operand order repaired and checked by reading, not compiled, not run on hardware
//
#include <hip/hip_runtime.h>
#include <hip/hip_bf16.h>

// ---------------------------------------------------------------------------
// FBP pipeline for MI455X (gfx1250, wave32):
//   1) circle-interp  : features(16,1,8,128,128) -> proj f16 (128,1,8,128,128)
//   2) ramp filter    : circular conv == circulant GEMM via v_wmma_f32_16x16x32_f16
//   3) backprojection : per-(c,h) slice staged in LDS (64KiB) via async-to-LDS,
//                       bilinear gather over 128 angles
// ---------------------------------------------------------------------------

typedef __attribute__((ext_vector_type(16))) _Float16 v16h;
typedef __attribute__((ext_vector_type(8)))  _Float16 v8h;
typedef __attribute__((ext_vector_type(8)))  float    v8f;

#define N_ANGLES 128
#define W        128
#define SZ       128
#define NCH      8          // C
#define NH       128        // H
#define SPATIAL  (NCH * NH * W)        // 131072 elements per angle
#define NROWS    (N_ANGLES * NCH * NH) // 131072 GEMM rows

// gfx1250 async memory->LDS path (ASYNCcnt-tracked), with portable fallback.
#if defined(__AMDGCN__) && __has_builtin(__builtin_amdgcn_global_load_async_to_lds_b128)
#define USE_ASYNC_LDS 1
// Parameter types per hipcc diagnostic: (v4i addrspace(1)*, v4i addrspace(3)*, Ii, Ii)
typedef int vi4 __attribute__((vector_size(16)));
typedef __attribute__((address_space(1))) vi4* glob_b128_t;
typedef __attribute__((address_space(3))) vi4* lds_b128_t;
#else
#define USE_ASYNC_LDS 0
#endif

// ---------------------------------------------------------------------------
// Step 0a: ramp-filter impulse response h[n] = (1/W) * sum_k |fftfreq(k)| cos(2*pi*k*n/W)
// ---------------------------------------------------------------------------
__global__ void ramp_h_kernel(float* __restrict__ hbuf) {
    const int n = threadIdx.x;   // 0..127
    float s = 0.0f;
    for (int k = 0; k < W; ++k) {
        const int  kk = (k <= W - k) ? k : (W - k);     // min(k, W-k)
        const float fa = (float)kk * (1.0f / (float)W); // |fftfreq|
        s += fa * cosf(6.28318530717958647692f * (float)(k * n) * (1.0f / (float)W));
    }
    hbuf[n] = s * (1.0f / (float)W);
}

// ---------------------------------------------------------------------------
// Step 0b: circulant matrix (B of the GEMM), stored K-contiguous per output col:
//   Hh[n][k] = h[(n - k) & 127]   (f16, 128x128 = 32 KiB)
// ---------------------------------------------------------------------------
__global__ void hh_kernel(const float* __restrict__ hbuf, _Float16* __restrict__ Hh) {
    const int idx = blockIdx.x * 256 + threadIdx.x;     // 16384 total
    const int n = idx >> 7;
    const int k = idx & 127;
    Hh[idx] = (_Float16)hbuf[(n - k) & 127];
}

// ---------------------------------------------------------------------------
// Step 1: circle interpolation along angle axis (16 wrap -> 128 angles), f32 -> f16
//   pos(a) = a/8 ; i0 = a>>3 ; w = (a&7)/8 ; stacked[16] == features[0]
// ---------------------------------------------------------------------------
__global__ void interp_kernel(const float* __restrict__ feat, _Float16* __restrict__ proj) {
    const int idx = blockIdx.x * 256 + threadIdx.x;     // 128 * 131072 total
    const int spatial = idx & (SPATIAL - 1);
    const int a  = idx >> 17;
    const int i0 = a >> 3;
    const float w = (float)(a & 7) * 0.125f;
    int i1 = i0 + 1;
    if (i1 >= 16) i1 = 0;                               // wrap: stacked[16] = features[0]
    const float f0 = feat[i0 * SPATIAL + spatial];
    const float f1 = feat[i1 * SPATIAL + spatial];
    proj[idx] = (_Float16)(f0 + w * (f1 - f0));
}

// ---------------------------------------------------------------------------
// Step 2: ramp filter as GEMM  filtered[r][n] = sum_k proj[r][k] * Hh[n][k]
//   M = 131072 rows, N = 128, K = 128.  One wave per 16x16 tile, 4 WMMAs (K=32 each).
//   A VGPR layout (16-bit A 16x32, ISA 7.12.2): lane half h, elems 0..7 -> K=8h+e,
//   elems 8..15 -> K=16+8h+(e-8).  B: same interleaved K order as A (rows of Hh).
// ---------------------------------------------------------------------------
__device__ inline v16h load_a_frag(const _Float16* __restrict__ p) {
    // p points at row base + k0 (k0 = 32*kk + 8*half); need [k0..k0+7] and [k0+16..k0+23]
    v8h lo = *(const v8h*)(p);
    v8h hi = *(const v8h*)(p + 16);
    v16h r;
#pragma unroll
    for (int i = 0; i < 8; ++i) { r[i] = lo[i]; r[i + 8] = hi[i]; }
    return r;
}

__global__ void filter_wmma_kernel(const _Float16* __restrict__ proj,
                                   const _Float16* __restrict__ Hh,
                                   float* __restrict__ filtered) {
    const int wave = threadIdx.x >> 5;      // 0..7 : column tile (8 * 16 = 128 cols)
    const int lane = threadIdx.x & 31;
    const int half = lane >> 4;
    const int l16  = lane & 15;
    const int rbase = blockIdx.x * 16;      // 8192 blocks * 16 rows = 131072
    const int row   = rbase + l16;          // A row handled by this lane
    const int ncol  = wave * 16 + l16;      // B column (row of Hh) / D column

    const _Float16* arow = proj + (size_t)row  * W;
    const _Float16* brow = Hh   + (size_t)ncol * W;

    v8f acc = {};
#pragma unroll
    for (int kk = 0; kk < 4; ++kk) {
        const int ka = kk * 32 + half * 8;      // interleaved A pattern
        v16h a = load_a_frag(arow + ka);
        v16h b = load_a_frag(brow + ka);        // same interleaved pattern for B
        acc = __builtin_amdgcn_wmma_f32_16x16x32_f16(
            /*neg_a=*/false, a, /*neg_b=*/false, b,
            /*c_mod=*/(short)0, acc, /*reuse_a=*/false, /*reuse_b=*/false);
    }

    // D layout: lane l16 = col, VGPR r -> row m = r + 8*half
#pragma unroll
    for (int r = 0; r < 8; ++r) {
        const int m = r + 8 * half;
        filtered[(size_t)(rbase + m) * W + ncol] = acc[r];
    }
}

// ---------------------------------------------------------------------------
// Step 3: backprojection.  One workgroup per (c,h) slice.
//   LDS: 128x128 f32 sinogram slice (64 KiB) + cos/sin tables (1 KiB), dynamic.
//   The slice is filled with GLOBAL_LOAD_ASYNC_TO_LDS_B128 (ASYNCcnt path) so the
//   fill overlaps the trig-table computation; each filtered element touches HBM once.
// ---------------------------------------------------------------------------
__global__ void backproject_kernel(const float* __restrict__ filtered,
                                   float* __restrict__ out) {
    extern __shared__ float lds[];
    float* sF = lds;                 // 16384 floats
    float* cs = lds + 16384;         // 128
    float* sn = lds + 16384 + 128;   // 128

    const int slice = blockIdx.x;    // c*128 + h, 1024 slices
    const int tid   = threadIdx.x;   // 256 threads = 8 waves

    // 16-byte granules: filtered row for angle a starts at (a*1024 + slice)*128
    for (int i = tid; i < (N_ANGLES * W) / 4; i += 256) {
        const int a = i >> 5;        // 32 float4 per row
        const int j = i & 31;
        const float* gsrc = filtered + ((size_t)a * 1024 + slice) * W + (j << 2);
        float*       ldst = sF + (a << 7) + (j << 2);
#if USE_ASYNC_LDS
        __builtin_amdgcn_global_load_async_to_lds_b128(
            (glob_b128_t)gsrc, (lds_b128_t)ldst,
            /*offset=*/0, /*cpol=*/0);
#else
        *(float4*)ldst = *(const float4*)gsrc;
#endif
    }
    if (tid < N_ANGLES) {
        const float th = (float)tid * (3.14159265358979323846f / (float)N_ANGLES);
        cs[tid] = cosf(th);
        sn[tid] = sinf(th);
    }
#if USE_ASYNC_LDS
    asm volatile("s_wait_asynccnt 0" ::: "memory");
#endif
    __syncthreads();

    for (int p = tid; p < SZ * SZ; p += 256) {
        const float xc = (float)(p >> 7) - 63.5f;
        const float yc = (float)(p & 127) - 63.5f;
        float acc = 0.0f;
#pragma unroll 4
        for (int a = 0; a < N_ANGLES; ++a) {
            const float u  = fmaf(xc, cs[a], fmaf(yc, sn[a], 63.5f));
            const float fl = floorf(u);
            const int   i0 = (int)fl;
            const float fr = u - fl;
            const float* row = sF + (a << 7);
            const float g0 = (i0 >= 0  && i0 < W)     ? row[i0]     : 0.0f;
            const float g1 = (i0 >= -1 && i0 < W - 1) ? row[i0 + 1] : 0.0f;
            acc = fmaf(g0, 1.0f - fr, fmaf(g1, fr, acc));
        }
        out[(size_t)slice * (SZ * SZ) + p] = acc * (3.14159265358979323846f / (float)N_ANGLES);
    }
}

// ---------------------------------------------------------------------------
// Host launcher
// ---------------------------------------------------------------------------
extern "C" void kernel_launch(void* const* d_in, const int* in_sizes, int n_in,
                              void* d_out, int out_size, void* d_ws, size_t ws_size,
                              hipStream_t stream) {
    (void)in_sizes; (void)n_in; (void)out_size; (void)ws_size;

    const float* feat = (const float*)d_in[0];
    float*       out  = (float*)d_out;
    char*        ws   = (char*)d_ws;

    // workspace layout (256-aligned offsets):
    float*     hbuf     = (float*)ws;                                   //   512 B
    _Float16*  Hh       = (_Float16*)(ws + 1024);                       //  32 KiB
    _Float16*  proj     = (_Float16*)(ws + 65536);                      //  32 MiB
    float*     filtered = (float*)(ws + 65536 + (size_t)NROWS * W * 2); //  64 MiB

    ramp_h_kernel<<<1, 128, 0, stream>>>(hbuf);
    hh_kernel<<<64, 256, 0, stream>>>(hbuf, Hh);
    interp_kernel<<<(N_ANGLES * SPATIAL) / 256, 256, 0, stream>>>(feat, proj);
    filter_wmma_kernel<<<NROWS / 16, 256, 0, stream>>>(proj, Hh, filtered);
    backproject_kernel<<<NCH * NH, 256, (16384 + 256) * sizeof(float), stream>>>(filtered, out);
}
